// LSTMDecoder_64201171140801
// MI455X (gfx1250) — matrix operand order repaired and checked by reading, not compiled, not run on hardware
//
#include <hip/hip_runtime.h>
#include <stdint.h>

#define DD 512
#define VV 32000
#define BB 64
#define TT 32

typedef __attribute__((ext_vector_type(16))) __bf16 v16bf;
typedef __attribute__((ext_vector_type(8)))  float  v8f;

__device__ __forceinline__ uint16_t f2bf(float f) {
  union { float f; uint32_t u; } c;
  c.f = f;
  uint32_t u = c.u;
  uint32_t r = 0x7FFFu + ((u >> 16) & 1u);   // round to nearest even
  return (uint16_t)((u + r) >> 16);
}

union FragAB { uint4 u[2]; v16bf v; };

// C[64, N] = A[64, K] (bf16, ld=lda) * Bsrc[N, K]^T (bf16, ld=ldb)  (+ bias[n])
// One wave owns the FULL M=64 column for one 16-wide N tile:
//   4 accumulators, 4 A fragments, 1 B fragment per K-step -> 4 WMMAs per B fetch.
// blockDim = 128 (4 waves = 4 N tiles), gridDim.x = N/64.
template <int K>
__global__ void wmma_gemm64_m4(const uint16_t* __restrict__ A, int lda,
                               const uint16_t* __restrict__ Bs, int ldb,
                               float* __restrict__ C, int ldc,
                               const float* __restrict__ bias) {
  const int lane  = threadIdx.x & 31;
  const int wave  = threadIdx.x >> 5;
  const int nBase = (blockIdx.x * 4 + wave) * 16;
  const int half  = lane >> 4;          // 0 or 1
  const int lrow  = lane & 15;

  // A frag (per M tile): lane -> M=lrow; elems 0-7 = K[k0..k0+7], 8-15 = K[k0+16..k0+23]
  const uint16_t* pa = A + (size_t)lrow * lda + half * 8;
  const size_t mStride = (size_t)16 * lda;
  // B frag: lane -> N=lrow; elems 0-7 = K[k0..k0+7], 8-15 = K[k0+16..k0+23]
  const uint16_t* pb = Bs + (size_t)(nBase + lrow) * ldb + half * 8;

  v8f acc0 = {}, acc1 = {}, acc2 = {}, acc3 = {};
#pragma unroll 2
  for (int k = 0; k < K; k += 32) {
    __builtin_prefetch(pb + k + 128, 0, 1);   // global_prefetch_b8: next B chunk
    FragAB b;
    b.u[0] = *(const uint4*)(pb + k);
    b.u[1] = *(const uint4*)(pb + k + 16);
    FragAB a0, a1, a2, a3;
    a0.u[0] = *(const uint4*)(pa + k);
    a0.u[1] = *(const uint4*)(pa + k + 16);
    a1.u[0] = *(const uint4*)(pa + mStride + k);
    a1.u[1] = *(const uint4*)(pa + mStride + k + 16);
    a2.u[0] = *(const uint4*)(pa + 2 * mStride + k);
    a2.u[1] = *(const uint4*)(pa + 2 * mStride + k + 16);
    a3.u[0] = *(const uint4*)(pa + 3 * mStride + k);
    a3.u[1] = *(const uint4*)(pa + 3 * mStride + k + 16);
    acc0 = __builtin_amdgcn_wmma_f32_16x16x32_bf16(false, a0.v, false, b.v,
                                                   (short)0, acc0, false, false);
    acc1 = __builtin_amdgcn_wmma_f32_16x16x32_bf16(false, a1.v, false, b.v,
                                                   (short)0, acc1, false, false);
    acc2 = __builtin_amdgcn_wmma_f32_16x16x32_bf16(false, a2.v, false, b.v,
                                                   (short)0, acc2, false, false);
    acc3 = __builtin_amdgcn_wmma_f32_16x16x32_bf16(false, a3.v, false, b.v,
                                                   (short)0, acc3, false, false);
  }

  // C/D layout: lane<16 -> N=lane, M=r ; lane>=16 -> N=lane-16, M=8+r
  const int n    = nBase + lrow;
  const float bv = bias ? bias[n] : 0.0f;
#define STORE_TILE(ACC, MT)                                            \
  {                                                                    \
    const int mOff = (MT) * 16 + half * 8;                             \
    _Pragma("unroll")                                                  \
    for (int r = 0; r < 8; ++r)                                        \
      C[(size_t)(mOff + r) * ldc + n] = ACC[r] + bv;                   \
  }
  STORE_TILE(acc0, 0)
  STORE_TILE(acc1, 1)
  STORE_TILE(acc2, 2)
  STORE_TILE(acc3, 3)
#undef STORE_TILE
}

// g[b,m] (no bias yet) -> gates -> h,c (fp32); h also packed bf16 into Xcat/Hcat.
__global__ void lstm_pointwise(const float* __restrict__ g,
                               const float* __restrict__ b_ih,
                               const float* __restrict__ b_hh,
                               float* __restrict__ h, float* __restrict__ c,
                               uint16_t* __restrict__ Xcat,
                               uint16_t* __restrict__ Hcat) {
  int idx = blockIdx.x * blockDim.x + threadIdx.x;  // 64*512
  int b = idx >> 9, d = idx & 511;
  const float* gb = g + (size_t)b * 4 * DD;
  float gi = gb[d]          + b_ih[d]          + b_hh[d];
  float gf = gb[DD + d]     + b_ih[DD + d]     + b_hh[DD + d];
  float gg = gb[2 * DD + d] + b_ih[2 * DD + d] + b_hh[2 * DD + d];
  float go = gb[3 * DD + d] + b_ih[3 * DD + d] + b_hh[3 * DD + d];
  float cp = c[idx];
  float si = 1.0f / (1.0f + __expf(-gi));
  float sf = 1.0f / (1.0f + __expf(-gf));
  float so = 1.0f / (1.0f + __expf(-go));
  float c2 = sf * cp + si * tanhf(gg);
  float h2 = so * tanhf(c2);
  c[idx] = c2;
  h[idx] = h2;
  uint16_t hb = f2bf(h2);
  Xcat[(size_t)b * 1024 + 512 + d] = hb;   // gate-GEMM input for next step
  Hcat[(size_t)b * 1024 + d]       = hb;   // z-GEMM input this step
}

// Outer-product attention. a[b,i,j]=exp(f_i h_j)/Z_j ; ctx_i = sum_j exp(f_i h_j) f_j / Z_j
__global__ void attention(const float* __restrict__ feats,
                          const float* __restrict__ h,
                          uint16_t* __restrict__ Hcat) {
  const int b = blockIdx.x, tid = threadIdx.x;   // 256 threads
  __shared__ float fs[DD], hs[DD], wsh[DD];
  fs[tid]       = feats[(size_t)b * DD + tid];
  fs[tid + 256] = feats[(size_t)b * DD + tid + 256];
  hs[tid]       = h[(size_t)b * DD + tid];
  hs[tid + 256] = h[(size_t)b * DD + tid + 256];
  __syncthreads();
  for (int j = tid; j < DD; j += 256) {
    float hj = hs[j], z = 0.0f;
    for (int i = 0; i < DD; ++i) z += __expf(fs[i] * hj);
    wsh[j] = fs[j] / z;
  }
  __syncthreads();
  for (int i = tid; i < DD; i += 256) {
    float fi = fs[i], ctx = 0.0f;
    for (int j = 0; j < DD; ++j) ctx += __expf(fi * hs[j]) * wsh[j];
    Hcat[(size_t)b * 1024 + 512 + i] = f2bf(ctx);
  }
}

// emb = tanh(LN(z + b_hc)); write bf16 into Xcat[:, 0:512]
__global__ void ln_tanh_emb(const float* __restrict__ z,
                            const float* __restrict__ b_hc,
                            const float* __restrict__ gamma,
                            const float* __restrict__ beta,
                            uint16_t* __restrict__ Xcat) {
  const int b = blockIdx.x, tid = threadIdx.x;   // 256 threads
  __shared__ float sh[DD];
  __shared__ float rs[256], rq[256];
  float x0 = z[(size_t)b * DD + tid]       + b_hc[tid];
  float x1 = z[(size_t)b * DD + tid + 256] + b_hc[tid + 256];
  sh[tid] = x0; sh[tid + 256] = x1;
  rs[tid] = x0 + x1;
  rq[tid] = x0 * x0 + x1 * x1;
  __syncthreads();
  for (int o = 128; o > 0; o >>= 1) {
    if (tid < o) { rs[tid] += rs[tid + o]; rq[tid] += rq[tid + o]; }
    __syncthreads();
  }
  float mean = rs[0] * (1.0f / DD);
  float var  = rq[0] * (1.0f / DD) - mean * mean;
  float rstd = rsqrtf(var + 1e-5f);
  float e0 = tanhf((sh[tid] - mean)       * rstd * gamma[tid]       + beta[tid]);
  float e1 = tanhf((sh[tid + 256] - mean) * rstd * gamma[tid + 256] + beta[tid + 256]);
  Xcat[(size_t)b * 1024 + tid]       = f2bf(e0);
  Xcat[(size_t)b * 1024 + tid + 256] = f2bf(e1);
}

__global__ void init_state(const float* __restrict__ feats,
                           const float* __restrict__ vocab_emb,
                           float* __restrict__ h, float* __restrict__ c,
                           uint16_t* __restrict__ Xcat) {
  int idx = blockIdx.x * blockDim.x + threadIdx.x;  // 64*512
  int b = idx >> 9, d = idx & 511;
  float f = feats[idx];
  h[idx] = f;
  c[idx] = f;
  Xcat[(size_t)b * 1024 + 512 + d] = f2bf(f);            // h0 = feats
  Xcat[(size_t)b * 1024 + d]       = f2bf(vocab_emb[d]); // emb0 = vocab_emb[0]
}

__global__ void conv_bf16(const float* __restrict__ src, uint16_t* __restrict__ dst, int n) {
  int i = blockIdx.x * blockDim.x + threadIdx.x;
  if (i < n) dst[i] = f2bf(src[i]);
}

// Wk[m, k] = k<512 ? W_ih[m,k] : W_hh[m,k-512]   (m<2048, k<1024), bf16
__global__ void pack_wk(const float* __restrict__ W_ih, const float* __restrict__ W_hh,
                        uint16_t* __restrict__ Wk) {
  int idx = blockIdx.x * blockDim.x + threadIdx.x;  // 2048*1024
  int m = idx >> 10, k = idx & 1023;
  float v = (k < 512) ? W_ih[(size_t)m * 512 + k] : W_hh[(size_t)m * 512 + (k - 512)];
  Wk[idx] = f2bf(v);
}

extern "C" void kernel_launch(void* const* d_in, const int* in_sizes, int n_in,
                              void* d_out, int out_size, void* d_ws, size_t ws_size,
                              hipStream_t stream) {
  (void)in_sizes; (void)n_in; (void)out_size; (void)ws_size;
  const float* feats = (const float*)d_in[0];
  const float* W_ih  = (const float*)d_in[1];
  const float* W_hh  = (const float*)d_in[2];
  const float* b_ih  = (const float*)d_in[3];
  const float* b_hh  = (const float*)d_in[4];
  const float* W_hc  = (const float*)d_in[5];
  const float* b_hc  = (const float*)d_in[6];
  const float* ln_g  = (const float*)d_in[7];
  const float* ln_b  = (const float*)d_in[8];
  const float* vocab = (const float*)d_in[9];
  const float* vbias = (const float*)d_in[10];
  float* out = (float*)d_out;

  char* p = (char*)d_ws;
  auto alloc = [&](size_t bytes) -> char* {
    char* r = p;
    p += (bytes + 255) & ~(size_t)255;
    return r;
  };
  uint16_t* vocab_bf = (uint16_t*)alloc((size_t)VV * DD * 2);     // 32.8 MB
  uint16_t* Wk       = (uint16_t*)alloc((size_t)4 * DD * 2 * DD * 2);
  uint16_t* Whc      = (uint16_t*)alloc((size_t)DD * 2 * DD * 2);
  uint16_t* Xcat     = (uint16_t*)alloc((size_t)BB * 2 * DD * 2); // [emb | h] bf16
  uint16_t* Hcat     = (uint16_t*)alloc((size_t)BB * 2 * DD * 2); // [h | ctx] bf16
  float*    hbuf     = (float*)alloc((size_t)BB * DD * 4);
  float*    cbuf     = (float*)alloc((size_t)BB * DD * 4);
  float*    gbuf     = (float*)alloc((size_t)BB * 4 * DD * 4);
  float*    zbuf     = (float*)alloc((size_t)BB * DD * 4);

  // One-time weight conversions (deterministic; rerun every call)
  conv_bf16<<<(VV * DD + 255) / 256, 256, 0, stream>>>(vocab, vocab_bf, VV * DD);
  conv_bf16<<<(DD * 2 * DD + 255) / 256, 256, 0, stream>>>(W_hc, Whc, DD * 2 * DD);
  pack_wk<<<(4 * DD * 2 * DD + 255) / 256, 256, 0, stream>>>(W_ih, W_hh, Wk);
  init_state<<<(BB * DD + 255) / 256, 256, 0, stream>>>(feats, vocab, hbuf, cbuf, Xcat);

  for (int t = 0; t < TT; ++t) {
    // gates: [64,1024]x[1024,2048]
    wmma_gemm64_m4<1024><<<4 * DD / 64, 128, 0, stream>>>(Xcat, 1024, Wk, 1024,
                                                          gbuf, 4 * DD, nullptr);
    lstm_pointwise<<<BB * DD / 256, 256, 0, stream>>>(gbuf, b_ih, b_hh, hbuf, cbuf,
                                                      Xcat, Hcat);
    attention<<<BB, 256, 0, stream>>>(feats, hbuf, Hcat);
    // z: [64,1024]x[1024,512]
    wmma_gemm64_m4<1024><<<DD / 64, 128, 0, stream>>>(Hcat, 1024, Whc, 1024,
                                                      zbuf, DD, nullptr);
    ln_tanh_emb<<<BB, 256, 0, stream>>>(zbuf, b_hc, ln_g, ln_b, Xcat);
    // logits: [64,512]x[512,32000]  (A = Xcat[:, 0:512], lda=1024)
    wmma_gemm64_m4<512><<<VV / 64, 128, 0, stream>>>(Xcat, 1024, vocab_bf, 512,
                                                     out + (size_t)t * BB * VV, VV, vbias);
  }
}
